// LSHAttention_18133351923835
// MI455X (gfx1250) — compile-verified, matrix operand order checked by reading, not run on hardware
//
#include <hip/hip_runtime.h>
#include <hip/hip_bf16.h>

typedef __attribute__((ext_vector_type(16))) _Float16 v16h;
typedef __attribute__((ext_vector_type(8)))  float    v8f;

#define B_      16
#define S_      4096
#define D_      64
#define H_      8
#define NB_     64          // n_buckets per hash round
#define NBUCK_G 512         // H_ * NB_ distinct bucket values
#define NTOK    32768       // H_ * S_ items per batch
#define NCHUNKS 512         // attention chunks per batch
#define CHN     64          // queries per attention chunk
#define SORT_CH   64        // histogram chunks per batch
#define SORT_CHSZ 512       // items per histogram chunk

#define QS      72          // padded row stride (halfs) for 64-wide tiles
#define PS      136         // padded row stride (halfs) for 128-wide tiles

// ---------------------------------------------------------------- zero output
__global__ void k_zero(float* __restrict__ p, int n) {
  int i = blockIdx.x * 256 + threadIdx.x;
  if (i < n) p[i] = 0.f;
}

// ---------------------------------------------------------------- LSH hashing
// rotated = qk[b,t,:] . rotations[b,:,h,i]; bucket = argmax([r,-r]) + h*NB_
__global__ void k_hash(const float* __restrict__ qk, const float* __restrict__ rot,
                       int* __restrict__ buckets) {
  __shared__ float srot[D_][32];
  int bh = blockIdx.y;
  int b = bh >> 3, h = bh & 7;
  int tid = threadIdx.x;
  for (int j = tid; j < D_ * 32; j += 256) {
    int f = j >> 5, i = j & 31;
    srot[f][i] = rot[(((size_t)b * D_ + f) * H_ + h) * 32 + i];
  }
  __syncthreads();
  int t = blockIdx.x * 256 + tid;
  const float4* qrow = (const float4*)(qk + ((size_t)b * S_ + t) * D_);
  float acc[32];
#pragma unroll
  for (int i = 0; i < 32; ++i) acc[i] = 0.f;
#pragma unroll 4
  for (int f4 = 0; f4 < 16; ++f4) {
    float4 q4 = qrow[f4];
#pragma unroll
    for (int i = 0; i < 32; ++i) {
      acc[i] += q4.x * srot[4 * f4 + 0][i];
      acc[i] += q4.y * srot[4 * f4 + 1][i];
      acc[i] += q4.z * srot[4 * f4 + 2][i];
      acc[i] += q4.w * srot[4 * f4 + 3][i];
    }
  }
  // argmax over concat([r, -r]) with first-occurrence tie-breaking
  float vmax = -1e30f, vmin = 1e30f;
  int imax = 0, imin = 0;
#pragma unroll
  for (int i = 0; i < 32; ++i) {
    if (acc[i] > vmax) { vmax = acc[i]; imax = i; }
    if (acc[i] < vmin) { vmin = acc[i]; imin = i; }
  }
  int bk = (vmax >= -vmin) ? imax : (32 + imin);
  buckets[(size_t)b * NTOK + h * S_ + t] = bk + h * NB_;
}

// ------------------------------------------------- counting sort: histograms
__global__ void k_hist(const int* __restrict__ buckets, int* __restrict__ counts) {
  __shared__ int hist[NBUCK_G];
  int b = blockIdx.x, ch = blockIdx.y, tid = threadIdx.x;
  hist[tid] = 0;
  __syncthreads();
  int v = buckets[(size_t)b * NTOK + ch * SORT_CHSZ + tid];
  atomicAdd(&hist[v], 1);
  __syncthreads();
  counts[((size_t)b * SORT_CH + ch) * NBUCK_G + tid] = hist[tid];
}

// ------------------------- counting sort: bucket starts + per-chunk bases
__global__ void k_scan(int* __restrict__ counts) {
  __shared__ int sd[NBUCK_G];
  int b = blockIdx.x, v = threadIdx.x;
  int total = 0;
  for (int ch = 0; ch < SORT_CH; ++ch)
    total += counts[((size_t)b * SORT_CH + ch) * NBUCK_G + v];
  sd[v] = total;
  __syncthreads();
  for (int off = 1; off < NBUCK_G; off <<= 1) {
    int x = (v >= off) ? sd[v - off] : 0;
    __syncthreads();
    sd[v] += x;
    __syncthreads();
  }
  int run = sd[v] - total;   // exclusive prefix = global bucket start
  for (int ch = 0; ch < SORT_CH; ++ch) {
    size_t idx = ((size_t)b * SORT_CH + ch) * NBUCK_G + v;
    int c = counts[idx];
    counts[idx] = run;
    run += c;
  }
}

// ------------------------------------------------- counting sort: scatter
__global__ void k_scatter(const int* __restrict__ buckets, const int* __restrict__ counts,
                          int* __restrict__ sortidx) {
  __shared__ int lb[SORT_CHSZ];
  int b = blockIdx.x, ch = blockIdx.y, tid = threadIdx.x;
  int g = ch * SORT_CHSZ + tid;
  int myb = buckets[(size_t)b * NTOK + g];
  lb[tid] = myb;
  __syncthreads();
  int rank = 0;
  for (int j = 0; j < tid; ++j) rank += (lb[j] == myb) ? 1 : 0;
  int pos = counts[((size_t)b * SORT_CH + ch) * NBUCK_G + myb] + rank;
  sortidx[(size_t)b * NTOK + pos] = g;   // stable: rank within chunk by index
}

// ------- WMMA fragment builder (ISA 7.12.2 wave32 16-bit A/B layout)
// A 16x32 (rows row0.., k0..k0+31) from row-major LDS; the B 32x16 layout
// is the mirror (N across lanes), so B of M^T uses the same loader on M.
__device__ inline v16h frag_rowmaj(const _Float16* base, int row0, int stride,
                                   int k0, int lane) {
  v16h f;
  const _Float16* rp = base + (row0 + (lane & 15)) * stride + k0
                            + (((lane >> 4) & 1) << 3);
#pragma unroll
  for (int v = 0; v < 8; ++v) {
    int k = ((v >> 2) << 4) + ((v & 3) << 1);
    f[2 * v]     = rp[k];
    f[2 * v + 1] = rp[k + 1];
  }
  return f;
}

// ------------------------------------------------------- chunked attention
// PASS2=false: compute per-row LSE only.  PASS2=true: recompute dots, fold
// round weight into p, p@V via WMMA, atomic-accumulate into out.
template <bool PASS2>
__global__ void k_attn(const float* __restrict__ qk, const float* __restrict__ vin,
                       const int* __restrict__ sortidx,
                       float* __restrict__ lse_u, const float* __restrict__ wgt,
                       float* __restrict__ out) {
  __shared__ _Float16 sq[CHN * QS];        // 64x64 queries (f16), padded
  __shared__ _Float16 sk[2 * CHN * QS];    // 128x64 normalized keys, padded
  __shared__ _Float16 svt[D_ * PS];        // V transposed: [n][k] 64x128
  __shared__ _Float16 sp[CHN * PS];        // 64x128 weighted probabilities
  __shared__ int sqt[CHN], skt[2 * CHN], sg[CHN];

  int c = blockIdx.x, b = blockIdx.y;
  int cprev = (c + NCHUNKS - 1) & (NCHUNKS - 1);   // look-one-back chunk
  int kr = threadIdx.x;                            // 0..127 = key row

  int g = (kr < CHN) ? sortidx[(size_t)b * NTOK + c * CHN + kr]
                     : sortidx[(size_t)b * NTOK + cprev * CHN + (kr - CHN)];
  int t = g & (S_ - 1);

  const float4* qrow = (const float4*)(qk + ((size_t)b * S_ + t) * D_);
  float ss = 0.f;
#pragma unroll
  for (int i = 0; i < 16; ++i) {
    float4 q4 = qrow[i];
    ss += q4.x * q4.x + q4.y * q4.y + q4.z * q4.z + q4.w * q4.w;
  }
  float inv = 1.f / fmaxf(sqrtf(ss), 1e-12f);
#pragma unroll
  for (int i = 0; i < 16; ++i) {
    float4 q4 = qrow[i];
    _Float16* kd = &sk[kr * QS + 4 * i];
    kd[0] = (_Float16)(q4.x * inv); kd[1] = (_Float16)(q4.y * inv);
    kd[2] = (_Float16)(q4.z * inv); kd[3] = (_Float16)(q4.w * inv);
    if (kr < CHN) {
      _Float16* qd = &sq[kr * QS + 4 * i];
      qd[0] = (_Float16)q4.x; qd[1] = (_Float16)q4.y;
      qd[2] = (_Float16)q4.z; qd[3] = (_Float16)q4.w;
    }
  }
  if (PASS2) {
    // stage V transposed: row kr of V becomes column kr of svt
    const float4* vrow = (const float4*)(vin + ((size_t)b * S_ + t) * D_);
#pragma unroll
    for (int i = 0; i < 16; ++i) {
      float4 v4 = vrow[i];
      svt[(4 * i + 0) * PS + kr] = (_Float16)v4.x;
      svt[(4 * i + 1) * PS + kr] = (_Float16)v4.y;
      svt[(4 * i + 2) * PS + kr] = (_Float16)v4.z;
      svt[(4 * i + 3) * PS + kr] = (_Float16)v4.w;
    }
  }
  skt[kr] = t;
  if (kr < CHN) { sqt[kr] = t; sg[kr] = g; }
  __syncthreads();

  int wave = kr >> 5, lane = kr & 31;
  int m0 = wave * 16;                         // each wave owns 16 query rows
  int hs = ((lane >> 4) & 1) << 3;            // row half offset (0 or 8)
  int ncol = lane & 15;

  v16h a0 = frag_rowmaj(sq, m0, QS, 0, lane);
  v16h a1 = frag_rowmaj(sq, m0, QS, 32, lane);

  v8f acc[8];
#pragma unroll
  for (int ct = 0; ct < 8; ++ct) {            // dots: 16x128 strip, K=64
    v16h b0 = frag_rowmaj(sk, ct * 16, QS, 0, lane);   // B = K^T (mirror)
    v16h b1 = frag_rowmaj(sk, ct * 16, QS, 32, lane);
    v8f ca = {};
    ca = __builtin_amdgcn_wmma_f32_16x16x32_f16(false, a0, false, b0,
                                                (short)0, ca, false, false);
    ca = __builtin_amdgcn_wmma_f32_16x16x32_f16(false, a1, false, b1,
                                                (short)0, ca, false, false);
    acc[ct] = ca;
  }

  // hoisted token ids for self-masking
  int tq[8], tk[8];
#pragma unroll
  for (int r = 0; r < 8; ++r) tq[r] = sqt[m0 + hs + r];
#pragma unroll
  for (int ct = 0; ct < 8; ++ct) tk[ct] = skt[ct * 16 + ncol];

  float rmax[8];
#pragma unroll
  for (int r = 0; r < 8; ++r) rmax[r] = -1e30f;
#pragma unroll
  for (int ct = 0; ct < 8; ++ct)
#pragma unroll
    for (int r = 0; r < 8; ++r) {
      float x = acc[ct][r] * 0.125f;          // d^-0.5
      if (tq[r] == tk[ct]) x = -50000.0f;     // self mask
      acc[ct][r] = x;
      rmax[r] = fmaxf(rmax[r], x);
    }
#pragma unroll
  for (int r = 0; r < 8; ++r)
#pragma unroll
    for (int off = 8; off >= 1; off >>= 1)    // stays within 16-lane half
      rmax[r] = fmaxf(rmax[r], __shfl_xor(rmax[r], off, 32));
  float rsum[8];
#pragma unroll
  for (int r = 0; r < 8; ++r) rsum[r] = 0.f;
#pragma unroll
  for (int ct = 0; ct < 8; ++ct)
#pragma unroll
    for (int r = 0; r < 8; ++r) rsum[r] += __expf(acc[ct][r] - rmax[r]);
#pragma unroll
  for (int r = 0; r < 8; ++r)
#pragma unroll
    for (int off = 8; off >= 1; off >>= 1)
      rsum[r] += __shfl_xor(rsum[r], off, 32);
  float lse[8];
#pragma unroll
  for (int r = 0; r < 8; ++r) lse[r] = rmax[r] + __logf(rsum[r]);

  if (!PASS2) {
    if (ncol == 0) {
#pragma unroll
      for (int r = 0; r < 8; ++r)
        lse_u[(size_t)b * NTOK + sg[m0 + hs + r]] = lse[r];
    }
    return;
  }

  // fold cross-round weight into p, stage in LDS
  float wrow[8];
#pragma unroll
  for (int r = 0; r < 8; ++r)
    wrow[r] = wgt[(size_t)b * NTOK + sg[m0 + hs + r]];
#pragma unroll
  for (int ct = 0; ct < 8; ++ct)
#pragma unroll
    for (int r = 0; r < 8; ++r)
      sp[(m0 + hs + r) * PS + ct * 16 + ncol] =
          (_Float16)(__expf(acc[ct][r] - lse[r]) * wrow[r]);
  __syncthreads();

  // out strip (16x64) = p(16x128) @ V(128x64); B-frag of V from svt rows
  v8f oacc[4];
#pragma unroll
  for (int cn = 0; cn < 4; ++cn) { v8f z = {}; oacc[cn] = z; }
#pragma unroll
  for (int kc = 0; kc < 4; ++kc) {
    v16h pa = frag_rowmaj(sp, m0, PS, kc * 32, lane);
#pragma unroll
    for (int cn = 0; cn < 4; ++cn) {
      v16h vb = frag_rowmaj(svt, cn * 16, PS, kc * 32, lane);
      oacc[cn] = __builtin_amdgcn_wmma_f32_16x16x32_f16(false, pa, false, vb,
                                                        (short)0, oacc[cn],
                                                        false, false);
    }
  }
#pragma unroll
  for (int cn = 0; cn < 4; ++cn)
#pragma unroll
    for (int r = 0; r < 8; ++r) {
      atomicAdd(&out[((size_t)b * S_ + tq[r]) * D_ + cn * 16 + ncol],
                oacc[cn][r]);
    }
}

// -------------------------------------------- cross-round softmax weights
__global__ void k_weights(const float* __restrict__ lse_u, float* __restrict__ wgt) {
  int idx = blockIdx.x * 256 + threadIdx.x;   // b*S_ + t
  int b = idx >> 12, t = idx & (S_ - 1);
  float l[H_];
  float M = -1e30f;
#pragma unroll
  for (int h = 0; h < H_; ++h) {
    l[h] = lse_u[(size_t)b * NTOK + h * S_ + t];
    M = fmaxf(M, l[h]);
  }
  float Z = 0.f;
#pragma unroll
  for (int h = 0; h < H_; ++h) Z += __expf(l[h] - M);
  float iZ = 1.f / Z;
#pragma unroll
  for (int h = 0; h < H_; ++h)
    wgt[(size_t)b * NTOK + h * S_ + t] = __expf(l[h] - M) * iZ;
}

// ---------------------------------------------------------------- launcher
extern "C" void kernel_launch(void* const* d_in, const int* in_sizes, int n_in,
                              void* d_out, int out_size, void* d_ws, size_t ws_size,
                              hipStream_t stream) {
  const float* qk  = (const float*)d_in[0];
  const float* v   = (const float*)d_in[1];
  const float* rot = (const float*)d_in[2];
  float* out = (float*)d_out;

  char* ws = (char*)d_ws;                       // 10 MB total
  int*   buckets = (int*)(ws);                            // B*NTOK ints (2MB)
  int*   counts  = (int*)(ws + 2 * 1024 * 1024);          // B*64*512 ints (2MB)
  int*   sortidx = (int*)(ws + 4 * 1024 * 1024);          // B*NTOK ints (2MB)
  float* lse_u   = (float*)(ws + 6 * 1024 * 1024);        // B*NTOK f32 (2MB)
  float* wgt     = (float*)(ws + 8 * 1024 * 1024);        // B*NTOK f32 (2MB)

  k_zero<<<(out_size + 255) / 256, 256, 0, stream>>>(out, out_size);
  k_hash<<<dim3(S_ / 256, B_ * H_), 256, 0, stream>>>(qk, rot, buckets);
  k_hist<<<dim3(B_, SORT_CH), NBUCK_G, 0, stream>>>(buckets, counts);
  k_scan<<<B_, NBUCK_G, 0, stream>>>(counts);
  k_scatter<<<dim3(B_, SORT_CH), SORT_CHSZ, 0, stream>>>(buckets, counts, sortidx);
  k_attn<false><<<dim3(NCHUNKS, B_), 128, 0, stream>>>(qk, v, sortidx, lse_u,
                                                       nullptr, nullptr);
  k_weights<<<(B_ * S_) / 256, 256, 0, stream>>>(lse_u, wgt);
  k_attn<true><<<dim3(NCHUNKS, B_), 128, 0, stream>>>(qk, v, sortidx, lse_u,
                                                      wgt, out);
}
